// NodeSinkhornPooling_16492674417226
// MI455X (gfx1250) — compile-verified, matrix operand order checked by reading, not run on hardware
//
#include <hip/hip_runtime.h>
#include <hip/hip_bf16.h>

// CDNA5 WMMA vector types (probe-confirmed signatures)
typedef __attribute__((ext_vector_type(16))) __bf16 v16bf;
typedef __attribute__((ext_vector_type(8)))  float  v8f;

#define S_DIM 128
#define K_DIM 256
#define D_DIM 256
#define N_ITER 100

#define EPS      0.09f                   // blur^p = 0.3^2
#define INV_EPS  (1.0f / 0.09f)
#define LOG_A   (-4.852030263919617f)    // -ln(128)
#define LOG_B   (-5.545177444479562f)    // -ln(256)
#define LOG_AB  (-10.397207708399180f)   // -ln(128*256)

__device__ __forceinline__ float wave_max32(float v) {
#pragma unroll
  for (int off = 16; off > 0; off >>= 1)
    v = fmaxf(v, __shfl_xor(v, off, 32));
  return v;
}
__device__ __forceinline__ float wave_sum32(float v) {
#pragma unroll
  for (int off = 16; off > 0; off >>= 1)
    v += __shfl_xor(v, off, 32);
  return v;
}

__global__ __launch_bounds__(256)
void node_sinkhorn_kernel(const float* __restrict__ samples,   // [N,128,256]
                          const float* __restrict__ codebook,  // [256,256]
                          float* __restrict__ out)             // [N,256]
{
  extern __shared__ float smem[];
  float* Cm = smem;                   // [128*256] cost matrix (LDS-resident)
  float* x2 = Cm + S_DIM * K_DIM;     // [128]
  float* y2 = x2 + S_DIM;             // [256]
  float* fv = y2 + K_DIM;             // [128]
  float* gv = fv + S_DIM;             // [256]
  float* hb = gv + K_DIM;             // [256]
  float* red = hb + K_DIM;            // [8]

  const int tid   = threadIdx.x;
  const int lane  = tid & 31;
  const int wave  = tid >> 5;          // 8 waves
  const int node  = blockIdx.x;
  const int m16   = lane & 15;
  const int halfw = lane >> 4;

  // ---------------- Phase 0: squared norms, init potentials ----------------
  {
    const float4* crow = reinterpret_cast<const float4*>(codebook + (size_t)tid * D_DIM);
    float s = 0.f;
#pragma unroll 8
    for (int j = 0; j < D_DIM / 4; ++j) {
      float4 v = crow[j];
      s += v.x * v.x + v.y * v.y + v.z * v.z + v.w * v.w;
    }
    y2[tid] = s;
    gv[tid] = 0.f;
    if (tid < S_DIM) {
      const float4* xrow =
          reinterpret_cast<const float4*>(samples + ((size_t)node * S_DIM + tid) * D_DIM);
      float t = 0.f;
#pragma unroll 8
      for (int j = 0; j < D_DIM / 4; ++j) {
        float4 v = xrow[j];
        t += v.x * v.x + v.y * v.y + v.z * v.z + v.w * v.w;
      }
      x2[tid] = t;
      fv[tid] = 0.f;
    }
  }
  __syncthreads();

  // ---------------- Phase 1: WMMA GEMM -> cost matrix in LDS ----------------
  // Wave w owns S-rows [16w, 16w+16). A-frag layout (16-bit A 16x32):
  // lane = half*16 + m, element i=(q*8+r) holds K = q*16 + half*8 + r.
  v16bf afrag[8];
  {
    const float* arow = samples + ((size_t)node * S_DIM + wave * 16 + m16) * D_DIM;
#pragma unroll
    for (int c = 0; c < 8; ++c)
#pragma unroll
      for (int q = 0; q < 2; ++q)
#pragma unroll
        for (int r = 0; r < 8; ++r)
          afrag[c][q * 8 + r] = (__bf16)arow[c * 32 + q * 16 + halfw * 8 + r];
  }
#pragma unroll 1
  for (int tk = 0; tk < 16; ++tk) {
    const float* brow = codebook + (size_t)(tk * 16 + m16) * D_DIM;
    v8f acc = {};
#pragma unroll
    for (int c = 0; c < 8; ++c) {
      v16bf bfrag;
#pragma unroll
      for (int q = 0; q < 2; ++q)
#pragma unroll
        for (int r = 0; r < 8; ++r)
          bfrag[q * 8 + r] = (__bf16)brow[c * 32 + q * 16 + halfw * 8 + r];
      acc = __builtin_amdgcn_wmma_f32_16x16x32_bf16(
          /*neg_a=*/false, afrag[c], /*neg_b=*/false, bfrag,
          /*c_mod=*/(short)0, acc, /*reuse_a=*/false, /*reuse_b=*/false);
    }
    // C/D layout: VGPR r, lane = half*16 + n  ->  M = half*8 + r, N = n.
    // Fuse cost transform: C = 0.5*(|x|^2 + |y|^2) - x.y
#pragma unroll
    for (int r = 0; r < 8; ++r) {
      int srow = wave * 16 + halfw * 8 + r;
      int kcol = tk * 16 + m16;
      Cm[srow * K_DIM + kcol] = 0.5f * (x2[srow] + y2[kcol]) - acc[r];
    }
  }
  __syncthreads();

  // ---------------- Phase 2: 100 log-domain Sinkhorn iterations -------------
#pragma unroll 1
  for (int it = 0; it < N_ITER; ++it) {
    // f[s] = -eps * ( log_b + logsumexp_k( (g[k]-C[s,k])/eps ) )
    // wave handles 16 rows; each lane covers 8 columns (stride-32, coalesced)
#pragma unroll 1
    for (int rr = 0; rr < 16; ++rr) {
      const int s = wave * 16 + rr;
      float z[8];
      float mx = -3.402823466e38f;
#pragma unroll
      for (int j = 0; j < 8; ++j) {
        const int k = lane + j * 32;
        z[j] = (gv[k] - Cm[s * K_DIM + k]) * INV_EPS;
        mx = fmaxf(mx, z[j]);
      }
      mx = wave_max32(mx);
      float sm = 0.f;
#pragma unroll
      for (int j = 0; j < 8; ++j) sm += __expf(z[j] - mx);
      sm = wave_sum32(sm);
      if (lane == 0) fv[s] = -EPS * (mx + __logf(sm) + LOG_B);
    }
    __syncthreads();

    // g[k] = -eps * ( log_a + logsumexp_s( (f[s]-C[s,k])/eps ) )
    // one column per thread; 32 lanes hit consecutive addresses per row
    {
      const int k = tid;
      float mx = -3.402823466e38f;
#pragma unroll 4
      for (int s = 0; s < S_DIM; ++s)
        mx = fmaxf(mx, (fv[s] - Cm[s * K_DIM + k]) * INV_EPS);
      float sm = 0.f;
#pragma unroll 4
      for (int s = 0; s < S_DIM; ++s)
        sm += __expf((fv[s] - Cm[s * K_DIM + k]) * INV_EPS - mx);
      gv[k] = -EPS * (mx + __logf(sm) + LOG_A);
    }
    __syncthreads();
  }

  // ---------------- Phase 3: histogram = column marginals, normalized -------
  {
    const int k = tid;
    float sm = 0.f;
#pragma unroll 4
    for (int s = 0; s < S_DIM; ++s)
      sm += __expf((fv[s] + gv[k] - Cm[s * K_DIM + k]) * INV_EPS + LOG_AB);
    hb[k] = sm;
    float ws = wave_sum32(sm);
    if (lane == 0) red[wave] = ws;
  }
  __syncthreads();
  {
    float total = 0.f;
#pragma unroll
    for (int w = 0; w < 8; ++w) total += red[w];
    out[(size_t)node * K_DIM + tid] = hb[tid] / total;
  }
}

extern "C" void kernel_launch(void* const* d_in, const int* in_sizes, int n_in,
                              void* d_out, int out_size, void* d_ws, size_t ws_size,
                              hipStream_t stream) {
  (void)n_in; (void)out_size; (void)d_ws; (void)ws_size;
  const float* samples  = (const float*)d_in[0];   // [N,128,256] f32
  const float* codebook = (const float*)d_in[1];   // [256,256]   f32
  float* out = (float*)d_out;                      // [N,256]     f32

  const int N = in_sizes[0] / (S_DIM * D_DIM);     // 2048
  const size_t shmem =
      (size_t)(S_DIM * K_DIM + S_DIM + K_DIM + S_DIM + K_DIM + K_DIM + 8) * sizeof(float);

  hipLaunchKernelGGL(node_sinkhorn_kernel, dim3(N), dim3(256), shmem, stream,
                     samples, codebook, out);
}
